// CausalSelfAttention_38104949850255
// MI455X (gfx1250) — matrix operand order repaired and checked by reading, not compiled, not run on hardware
//
#include <hip/hip_runtime.h>

// ---------------------------------------------------------------------------
// CausalSelfAttention for MI455X (gfx1250, wave32, WMMA, TDM).
// Pipeline: f32->bf16 preconvert -> qkv_proj (bf16 WMMA, TDM-staged W tiles)
//           -> norm+rope -> flash attention (bf16 WMMA, online softmax)
//           -> out_proj (bf16 WMMA, f32 out).
// ---------------------------------------------------------------------------

typedef __attribute__((ext_vector_type(16))) __bf16 v16bf;
typedef __attribute__((ext_vector_type(8)))  __bf16 v8bf;
typedef __attribute__((ext_vector_type(4)))  __bf16 v4bf;
typedef __attribute__((ext_vector_type(8)))  float  v8f;
typedef __attribute__((ext_vector_type(4)))  float  v4f;
typedef __attribute__((ext_vector_type(4)))  unsigned v4u;
typedef __attribute__((ext_vector_type(8)))  int    v8i;
typedef __attribute__((ext_vector_type(4)))  int    v4i;

static constexpr int Bsz = 2;
static constexpr int T   = 2048;
static constexpr int C   = 1536;
static constexpr int NH  = 16;
static constexpr int HD  = 96;     // head dim
static constexpr int N3  = 3 * C;  // 4608

static __device__ inline __bf16 f2bf(float f) { return (__bf16)f; }
static __device__ inline float  bf2f(__bf16 b) { return (float)b; }

static __device__ inline v8f wmma_bf16(v16bf a, v16bf b, v8f c) {
  // D = A(16x32) * B(32x16) + C, f32 accumulate
  return __builtin_amdgcn_wmma_f32_16x16x32_bf16(false, a, false, b,
                                                 (short)0, c, false, false);
}
static __device__ inline v8f vzero8() {
  v8f z;
#pragma unroll
  for (int i = 0; i < 8; ++i) z[i] = 0.0f;
  return z;
}
// concat two contiguous 8-element runs into one A-fragment
static __device__ inline v16bf cat8(v8bf lo, v8bf hi) {
  return __builtin_shufflevector(lo, hi, 0, 1, 2, 3, 4, 5, 6, 7,
                                         8, 9, 10, 11, 12, 13, 14, 15);
}
// A-fragment: per-lane contiguous run at [base, base+16)
static __device__ inline v16bf load_a_frag(const __bf16* base) {
  return cat8(*(const v8bf*)base, *(const v8bf*)(base + 8));
}

// ---------------------------------------------------------------------------
// Tensor Data Mover staging of a 64(n) x 32(k) bf16 tile (row-major in LDS).
// D# per CDNA5 ISA: group0 {count=1, lds_addr,
// global_addr[56:0], type=2}; group1 {data_size=2B, tensor_dim0=K,
// tensor_dim1=nrows, tile=32x64, stride0=K}.
// ---------------------------------------------------------------------------
#if __has_builtin(__builtin_amdgcn_tensor_load_to_lds)
#define HAVE_TDM 1
#if defined(__clang_major__) && __clang_major__ >= 23
#define TDM_SIX_ARGS 1
#else
#define TDM_SIX_ARGS 0
#endif
#else
#define HAVE_TDM 0
#endif

static __device__ inline void tdm_stage(const __bf16* gbase, unsigned lds_off,
                                        unsigned nrows_tensor) {
#if HAVE_TDM
  unsigned long long ga = (unsigned long long)(size_t)gbase;
  v4u g0 = { 1u,                                   // count=1 (valid user D#)
             lds_off,                              // lds_addr (bytes)
             (unsigned)(ga & 0xFFFFFFFFull),
             (unsigned)(((ga >> 32) & 0x01FFFFFFull) | 0x80000000u) }; // type=2
  const unsigned td0 = (unsigned)C;                // row length (elements)
  v8i g1;
  g1[0] = (int)0x00010000u;                        // data_size=1 -> 2 bytes
  g1[1] = (int)((td0 & 0xFFFFu) << 16);            // tensor_dim0[15:0]
  g1[2] = (int)((td0 >> 16) | ((nrows_tensor & 0xFFFFu) << 16));
  g1[3] = (int)(((nrows_tensor >> 16) & 0xFFFFu) | (32u << 16)); // tile_dim0=32
  g1[4] = 64;                                      // tile_dim1=64, tile_dim2=0
  g1[5] = (int)td0;                                // tensor_dim0_stride = K
  g1[6] = 0;
  g1[7] = 0;
  v4i z4 = { 0, 0, 0, 0 };
#if TDM_SIX_ARGS
  v8i z8 = { 0, 0, 0, 0, 0, 0, 0, 0 };
  __builtin_amdgcn_tensor_load_to_lds(g0, g1, z4, z4, z8, 0);
#else
  __builtin_amdgcn_tensor_load_to_lds(g0, g1, z4, z4, 0);
#endif
#else
  (void)gbase; (void)lds_off; (void)nrows_tensor;
#endif
}

// Stage one 64x32 bf16 W tile into lbuf[buf] (raw [64][32] row-major).
static __device__ inline void stage_tile(const __bf16* __restrict__ Wbf,
                                         int n0, int kc, int buf, int tid,
                                         unsigned nrows,
                                         __bf16 (*lbuf)[64][32]) {
#if HAVE_TDM
  if ((tid >> 5) == 0)  // single wave issues the DMA (EXEC ignored by TDM)
    tdm_stage(Wbf + (size_t)n0 * C + kc, (unsigned)(buf * 4096), nrows);
  (void)lbuf;
#else
  int n = tid >> 2, seg = tid & 3;  // 256 thr x 16B = 4KB tile
  *(v8bf*)&lbuf[buf][n][seg * 8] =
      *(const v8bf*)(Wbf + (size_t)(n0 + n) * C + kc + seg * 8);
  (void)nrows;
#endif
}
static __device__ inline void stage_wait(int tid) {
#if HAVE_TDM
  if ((tid >> 5) == 0) {
#if __has_builtin(__builtin_amdgcn_s_wait_tensorcnt)
    __builtin_amdgcn_s_wait_tensorcnt(0);
#else
    asm volatile("s_wait_tensorcnt 0" ::: "memory");
#endif
  }
#else
  (void)tid;
#endif
  __syncthreads();
  // Compiler-level clobber: the TDM DMA writes LDS behind the compiler's
  // back; force B-fragment ds_loads to stay in the loop and not fold.
  asm volatile("" ::: "memory");
}
// Mark the LDS tile as written so its loads are not folded to undef.
// Never executes ((int)blockIdx.x < 0 is not provable at compile time).
#define LDS_MARK_WRITTEN(lbuf)                                        \
  do {                                                                \
    if ((int)blockIdx.x < 0) {                                        \
      volatile __bf16* pv_ = &(lbuf)[0][0][0];                        \
      *pv_ = (__bf16)0.0f;                                            \
    }                                                                 \
  } while (0)

// ---------------------------------------------------------------------------
// Kernel 0: f32 -> bf16 bulk convert (n divisible by 4).
// ---------------------------------------------------------------------------
__global__ void cvt_bf16_kernel(const float* __restrict__ in,
                                __bf16* __restrict__ out) {
  size_t i = ((size_t)blockIdx.x * blockDim.x + threadIdx.x) * 4;
  v4f v = *(const v4f*)(in + i);
  v4bf o;
#pragma unroll
  for (int j = 0; j < 4; ++j) o[j] = f2bf(v[j]);
  *(v4bf*)(out + i) = o;
}

// ---------------------------------------------------------------------------
// Kernel 1: QKV projection. qkv = xbf @ Wbf^T (M=4096, K=C, N=3C), bf16 WMMA.
// Block 256 thr (8 waves): tile 256(M) x 64(N); wave = 32x64 (8 WMMA/chunk).
// W tiles TDM-staged + double-buffered in LDS (raw row-major).
// Scatter: q,k -> [B,H,T,D] bf16 ; v -> [B,H,D,T] bf16 (transposed).
// ---------------------------------------------------------------------------
__global__ void qkv_proj_kernel(const __bf16* __restrict__ Abf,
                                const __bf16* __restrict__ Wbf,
                                __bf16* __restrict__ qbuf,
                                __bf16* __restrict__ kbuf,
                                __bf16* __restrict__ vtbuf) {
  __shared__ __align__(64) __bf16 lbuf[2][64][32];
  const int tid = threadIdx.x, lane = tid & 31, wave = tid >> 5;
  const int hi = lane >> 4, l15 = lane & 15;
  const int n0 = blockIdx.x * 64;
  const int m0 = blockIdx.y * 256 + wave * 32;

  LDS_MARK_WRITTEN(lbuf);

  v8f acc[2][4];
#pragma unroll
  for (int u = 0; u < 2; ++u)
#pragma unroll
    for (int s = 0; s < 4; ++s) acc[u][s] = vzero8();

  stage_tile(Wbf, n0, 0, 0, tid, (unsigned)N3, lbuf);
  int buf = 0;
  for (int kc = 0; kc < C; kc += 32, buf ^= 1) {
    stage_wait(tid);
    if (kc + 32 < C) stage_tile(Wbf, n0, kc + 32, buf ^ 1, tid, (unsigned)N3, lbuf);

    v16bf af[2];
#pragma unroll
    for (int u = 0; u < 2; ++u)
      af[u] = load_a_frag(Abf + (size_t)(m0 + u * 16 + l15) * C + kc + hi * 16);
#pragma unroll
    for (int s = 0; s < 4; ++s) {
      v16bf bfr = *(const v16bf*)&lbuf[buf][s * 16 + l15][hi * 16];
#pragma unroll
      for (int u = 0; u < 2; ++u) acc[u][s] = wmma_bf16(af[u], bfr, acc[u][s]);
    }
  }

  // scatter (C/D layout: M = r + 8*hi, N = l15)
#pragma unroll
  for (int u = 0; u < 2; ++u)
#pragma unroll
    for (int s = 0; s < 4; ++s)
#pragma unroll
      for (int r = 0; r < 8; ++r) {
        int m = m0 + u * 16 + r + 8 * hi;
        int n = n0 + s * 16 + l15;
        int sec = n / C, c = n % C;
        int h = c / HD, d = c % HD;
        int b = m / T,  t = m % T;
        __bf16 bv = f2bf(acc[u][s][r]);
        size_t bh = (size_t)b * NH + h;
        if (sec == 0)      qbuf[(bh * T + t) * HD + d] = bv;
        else if (sec == 1) kbuf[(bh * T + t) * HD + d] = bv;
        else               vtbuf[(bh * HD + d) * T + t] = bv;
      }
}

// ---------------------------------------------------------------------------
// Kernel 2: qk RMS-norm + 3D RoPE (in place, [B,H,T,D] bf16).
// ---------------------------------------------------------------------------
__global__ void norm_rope_kernel(__bf16* __restrict__ qbuf,
                                 __bf16* __restrict__ kbuf,
                                 const float* __restrict__ coords,
                                 const int* __restrict__ ttype,
                                 const float* __restrict__ q_scale,
                                 const float* __restrict__ k_scale) {
  const int row = blockIdx.x;            // 0 .. B*H*T-1
  const int isk = blockIdx.y;
  const int t   = row % T;
  const int b   = (row / T) / NH;
  __bf16* buf     = (isk ? kbuf : qbuf) + (size_t)row * HD;
  const float* sc = isk ? k_scale : q_scale;
  const int d = threadIdx.x;             // 0..95

  __shared__ float rv[HD];
  __shared__ float s_inv;
  float v = bf2f(buf[d]);
  rv[d] = v;
  __syncthreads();
  if (threadIdx.x == 0) {
    float acc = 0.0f;
    for (int i = 0; i < HD; ++i) acc += rv[i] * rv[i];
    s_inv = rsqrtf(acc / (float)HD + 1e-6f);
  }
  __syncthreads();
  const float inv = s_inv;
  float val  = v * inv * sc[d];
  float outv = val;
  if (ttype[(size_t)b * T + t] > 0) {
    int a = d / 32, dd = d % 32;
    int i = (dd < 16) ? dd : dd - 16;
    float ang = coords[((size_t)b * T + t) * 3 + a] *
                powf(10000.0f, -(float)i / 16.0f);
    float cs = cosf(ang), sn = sinf(ang);
    if (dd < 16) {
      int p = a * 32 + dd + 16;
      outv = val * cs - (rv[p] * inv * sc[p]) * sn;
    } else {
      int p = a * 32 + dd - 16;
      outv = (rv[p] * inv * sc[p]) * sn + val * cs;
    }
  }
  buf[d] = f2bf(outv);
}

// ---------------------------------------------------------------------------
// Kernel 3: flash attention. One wave per (b,h, 16-query block); 32-key steps.
// 6 WMMAs for S, shfl_xor online softmax, P via per-wave LDS (C->A layout,
// explicit s_wait_dscnt), 6 WMMAs for O += P*V (contiguous Vt fragments).
// ---------------------------------------------------------------------------
__global__ void flash_attn_kernel(const __bf16* __restrict__ qbuf,
                                  const __bf16* __restrict__ kbuf,
                                  const __bf16* __restrict__ vtbuf,
                                  __bf16* __restrict__ obuf) {
  const int lane = threadIdx.x & 31;
  const int wave = threadIdx.x >> 5;
  const int hi = lane >> 4, l15 = lane & 15;
  const int gid  = blockIdx.x * 8 + wave;   // 0 .. B*H*(T/16)-1
  const int qblk = gid & 127;               // T/16 = 128
  const int bh   = gid >> 7;                // 0..31
  const int h = bh % NH, b = bh / NH;

  const __bf16* Q  = qbuf  + (size_t)bh * T * HD;
  const __bf16* K  = kbuf  + (size_t)bh * T * HD;
  const __bf16* Vt = vtbuf + (size_t)bh * HD * T;

  __shared__ __align__(64) __bf16 plds_all[8][16][32];
  __bf16 (*plds)[32] = plds_all[wave];

  v16bf qf[3];
#pragma unroll
  for (int c = 0; c < 3; ++c)
    qf[c] = load_a_frag(Q + (size_t)(qblk * 16 + l15) * HD + c * 32 + hi * 16);

  v8f o[6];
#pragma unroll
  for (int tdx = 0; tdx < 6; ++tdx) o[tdx] = vzero8();
  float rmax[8], rsum[8];
#pragma unroll
  for (int r = 0; r < 8; ++r) { rmax[r] = -__builtin_inff(); rsum[r] = 0.0f; }

  const float scale = 0.10206207262f;   // 1/sqrt(96)
  const int nk = qblk * 16 + 16;

  for (int kb0 = 0; kb0 < nk; kb0 += 32) {
    if (kb0 + 32 < nk)  // pull next key block toward L2
      __builtin_prefetch(K + (size_t)(kb0 + 32 + l15) * HD, 0, 1);

    // ---- S = Q K^T over two 16-key halves ----
    v8f s0 = vzero8(), s1 = vzero8();
#pragma unroll
    for (int c = 0; c < 3; ++c) {
      v16bf kf0 = *(const v16bf*)(K + (size_t)(kb0 + l15) * HD + c * 32 + hi * 16);
      v16bf kf1 = *(const v16bf*)(K + (size_t)(kb0 + 16 + l15) * HD + c * 32 + hi * 16);
      s0 = wmma_bf16(qf[c], kf0, s0);
      s1 = wmma_bf16(qf[c], kf1, s1);
    }

    // ---- online softmax (row m = qblk*16 + r + 8*hi) ----
    float corr[8];
#pragma unroll
    for (int r = 0; r < 8; ++r) {
      const int m = qblk * 16 + r + 8 * hi;
      float a0 = (kb0 + l15      > m) ? -__builtin_inff() : s0[r] * scale;
      float a1 = (kb0 + 16 + l15 > m) ? -__builtin_inff() : s1[r] * scale;
      float mx = fmaxf(a0, a1);
#pragma unroll
      for (int off = 1; off < 16; off <<= 1) mx = fmaxf(mx, __shfl_xor(mx, off, 32));
      const float nm = fmaxf(rmax[r], mx);
      corr[r] = __expf(rmax[r] - nm);
      const float p0 = __expf(a0 - nm), p1 = __expf(a1 - nm);
      float ps = p0 + p1;
#pragma unroll
      for (int off = 1; off < 16; off <<= 1) ps += __shfl_xor(ps, off, 32);
      rsum[r] = rsum[r] * corr[r] + ps;
      rmax[r] = nm;
      const int ml = r + 8 * hi;
      plds[ml][l15]      = f2bf(p0);
      plds[ml][16 + l15] = f2bf(p1);
    }
#pragma unroll
    for (int tdx = 0; tdx < 6; ++tdx)
#pragma unroll
      for (int r = 0; r < 8; ++r) o[tdx][r] *= corr[r];

    asm volatile("s_wait_dscnt 0" ::: "memory");  // same-wave LDS RAW

    v16bf pf = load_a_frag(&plds[l15][hi * 16]);

#pragma unroll
    for (int tdx = 0; tdx < 6; ++tdx) {
      v16bf vf = *(const v16bf*)(Vt + (size_t)(tdx * 16 + l15) * T + kb0 + hi * 16);
      o[tdx] = wmma_bf16(pf, vf, o[tdx]);
    }
  }

  float invs[8];
#pragma unroll
  for (int r = 0; r < 8; ++r) invs[r] = 1.0f / rsum[r];
#pragma unroll
  for (int tdx = 0; tdx < 6; ++tdx)
#pragma unroll
    for (int r = 0; r < 8; ++r) {
      size_t row = (size_t)b * T + qblk * 16 + r + 8 * hi;
      int col = h * HD + tdx * 16 + l15;
      obuf[row * C + col] = f2bf(o[tdx][r] * invs[r]);
    }
}

// ---------------------------------------------------------------------------
// Kernel 4: output projection. out = attn_bf @ Wout^T -> f32 d_out.
// Same tiling/staging as kernel 1.
// ---------------------------------------------------------------------------
__global__ void out_proj_kernel(const __bf16* __restrict__ Abf,
                                const __bf16* __restrict__ Wbf,
                                float* __restrict__ out) {
  __shared__ __align__(64) __bf16 lbuf[2][64][32];
  const int tid = threadIdx.x, lane = tid & 31, wave = tid >> 5;
  const int hi = lane >> 4, l15 = lane & 15;
  const int n0 = blockIdx.x * 64;
  const int m0 = blockIdx.y * 256 + wave * 32;

  LDS_MARK_WRITTEN(lbuf);

  v8f acc[2][4];
#pragma unroll
  for (int u = 0; u < 2; ++u)
#pragma unroll
    for (int s = 0; s < 4; ++s) acc[u][s] = vzero8();

  stage_tile(Wbf, n0, 0, 0, tid, (unsigned)C, lbuf);
  int buf = 0;
  for (int kc = 0; kc < C; kc += 32, buf ^= 1) {
    stage_wait(tid);
    if (kc + 32 < C) stage_tile(Wbf, n0, kc + 32, buf ^ 1, tid, (unsigned)C, lbuf);

    v16bf af[2];
#pragma unroll
    for (int u = 0; u < 2; ++u)
      af[u] = load_a_frag(Abf + (size_t)(m0 + u * 16 + l15) * C + kc + hi * 16);
#pragma unroll
    for (int s = 0; s < 4; ++s) {
      v16bf bfr = *(const v16bf*)&lbuf[buf][s * 16 + l15][hi * 16];
#pragma unroll
      for (int u = 0; u < 2; ++u) acc[u][s] = wmma_bf16(af[u], bfr, acc[u][s]);
    }
  }

#pragma unroll
  for (int u = 0; u < 2; ++u)
#pragma unroll
    for (int s = 0; s < 4; ++s)
#pragma unroll
      for (int r = 0; r < 8; ++r) {
        size_t m = (size_t)m0 + u * 16 + r + 8 * hi;
        int n = n0 + s * 16 + l15;
        out[m * C + n] = acc[u][s][r];
      }
}

// ---------------------------------------------------------------------------
extern "C" void kernel_launch(void* const* d_in, const int* in_sizes, int n_in,
                              void* d_out, int out_size, void* d_ws, size_t ws_size,
                              hipStream_t stream) {
  (void)in_sizes; (void)n_in; (void)out_size; (void)ws_size;
  const float* x       = (const float*)d_in[0];
  const float* coords  = (const float*)d_in[1];
  const int*   ttype   = (const int*)d_in[2];
  const float* W_qkv   = (const float*)d_in[3];
  const float* W_out   = (const float*)d_in[4];
  const float* q_scale = (const float*)d_in[5];
  const float* k_scale = (const float*)d_in[6];
  float* out = (float*)d_out;

  const size_t nX  = (size_t)Bsz * T * C;   // 6,291,456
  const size_t nWq = (size_t)N3 * C;        // 7,077,888
  const size_t nWo = (size_t)C * C;         // 2,359,296
  const size_t nQ  = (size_t)Bsz * NH * T * HD;  // 6,291,456

  __bf16* xbf  = (__bf16*)d_ws;
  __bf16* wqb  = xbf + nX;
  __bf16* wob  = wqb + nWq;
  __bf16* qb   = wob + nWo;
  __bf16* kb   = qb  + nQ;
  __bf16* vtb  = kb  + nQ;
  __bf16* ob   = vtb + nQ;   // [B*T, C] bf16

  // 0) bulk f32 -> bf16
  cvt_bf16_kernel<<<dim3((unsigned)(nX  / 4 / 256)), 256, 0, stream>>>(x, xbf);
  cvt_bf16_kernel<<<dim3((unsigned)(nWq / 4 / 256)), 256, 0, stream>>>(W_qkv, wqb);
  cvt_bf16_kernel<<<dim3((unsigned)(nWo / 4 / 256)), 256, 0, stream>>>(W_out, wob);

  // 1) QKV projection
  qkv_proj_kernel<<<dim3(N3 / 64, (Bsz * T) / 256), 256, 0, stream>>>(
      xbf, wqb, qb, kb, vtb);
  // 2) norm + rope
  norm_rope_kernel<<<dim3(Bsz * NH * T, 2), HD, 0, stream>>>(
      qb, kb, coords, ttype, q_scale, k_scale);
  // 3) flash attention
  flash_attn_kernel<<<dim3((Bsz * NH * (T / 16)) / 8), 256, 0, stream>>>(
      qb, kb, vtb, ob);
  // 4) output projection
  out_proj_kernel<<<dim3(C / 64, (Bsz * T) / 256), 256, 0, stream>>>(
      ob, wob, out);
}
